// SelfAttentionSingleHead_71193377898909
// MI455X (gfx1250) — matrix operand order repaired and checked by reading, not compiled, not run on hardware
//
#include <hip/hip_runtime.h>
#include <hip/hip_bf16.h>
#include <math.h>

typedef __attribute__((ext_vector_type(16))) _Float16 v16h;
typedef __attribute__((ext_vector_type(8)))  _Float16 v8h;
typedef __attribute__((ext_vector_type(8)))  float    v8f;

#define B_SZ 4
#define T_SZ 2048
#define C_SZ 1024              // n_embd == d_k == d_v
#define M_TOTAL (B_SZ * T_SZ)  // 8192 flattened rows

// ---------- helpers ----------
__device__ inline v16h mk16(v8h lo, v8h hi) {
  v16h r;
#pragma unroll
  for (int i = 0; i < 8; ++i) { r[i] = lo[i]; r[i + 8] = hi[i]; }
  return r;
}
__device__ inline v16h ld16(const _Float16* p) {  // 16 contiguous f16 (16B-aligned)
  return mk16(*(const v8h*)p, *(const v8h*)(p + 8));
}
__device__ inline v8f wmma_f16(v16h a, v16h b, v8f c) {
  return __builtin_amdgcn_wmma_f32_16x16x32_f16(false, a, false, b, (short)0, c,
                                                false, false);
}

// ---------- stage 0: fp32 -> fp16 ----------
__global__ void cvt_f32_to_f16(const float* __restrict__ src,
                               _Float16* __restrict__ dst, int n) {
  int i = blockIdx.x * blockDim.x + threadIdx.x;
  int stride = gridDim.x * blockDim.x;
  for (; i < n; i += stride) dst[i] = (_Float16)src[i];
}

// ---------- stage 1: fused QKV projection GEMM (double-buffered) ----------
// A = xh [8192 x 1024] f16 row-major, W selected by blockIdx.z (Wq/Wk/Wv),
// outputs: z==0 -> qh row-major, z==1 -> kh row-major, z==2 -> vT[b][d][t].
__global__ __launch_bounds__(256)
void qkv_gemm(const _Float16* __restrict__ xh, const _Float16* __restrict__ Wh,
              _Float16* __restrict__ qh, _Float16* __restrict__ kh,
              _Float16* __restrict__ vT) {
  __shared__ _Float16 As[2][128][40];   // [buf][m][k], padded
  __shared__ _Float16 Bt[2][128][40];   // [buf][n][k] transposed, padded

  const int tid  = threadIdx.x;
  const int wave = tid >> 5, lane = tid & 31;
  const int nl = lane & 15, half = lane >> 4;
  const int m0 = blockIdx.x * 128;
  const int n0 = blockIdx.y * 128;
  const int z  = blockIdx.z;
  const _Float16* Wl = Wh + (size_t)z * C_SZ * C_SZ;
  _Float16* dst = (z == 0) ? qh : (z == 1) ? kh : vT;
  const int tmode = (z == 2);

  const int msub = (wave & 3) * 32;   // 4 M-groups of 32
  const int nsub = (wave >> 2) * 64;  // 2 N-groups of 64

  // one global->LDS staging step (A tile 128x32, B tile transposed 32x128)
  auto stage = [&](int bi, int k0) {
    {
      const int r = tid >> 1, ch = (tid & 1) * 16;
      const _Float16* p = xh + (size_t)(m0 + r) * C_SZ + k0 + ch;
      v8h lo = *(const v8h*)p, hi = *(const v8h*)(p + 8);
      *(v8h*)&As[bi][r][ch]     = lo;
      *(v8h*)&As[bi][r][ch + 8] = hi;
    }
    {
      const int kr = tid >> 3, nc = (tid & 7) * 16;
      const _Float16* p = Wl + (size_t)(k0 + kr) * C_SZ + n0 + nc;
      v8h lo = *(const v8h*)p, hi = *(const v8h*)(p + 8);
#pragma unroll
      for (int i = 0; i < 8; ++i) {
        Bt[bi][nc + i][kr]     = lo[i];
        Bt[bi][nc + 8 + i][kr] = hi[i];
      }
    }
  };

  v8f acc[2][4] = {};
  const int KSTEPS = C_SZ / 32;  // 32

  stage(0, 0);
  for (int kk = 0; kk < KSTEPS; ++kk) {
    __syncthreads();
    if (kk + 1 < KSTEPS) stage((kk + 1) & 1, (kk + 1) * 32);  // overlap next tile
    const int bi = kk & 1;

    v16h afr[2], bfr[4];
#pragma unroll
    for (int am = 0; am < 2; ++am) {
      const _Float16* ap = &As[bi][msub + am * 16 + nl][0];
      afr[am] = mk16(*(const v8h*)(ap + half * 16),
                     *(const v8h*)(ap + half * 16 + 8));
    }
#pragma unroll
    for (int bn = 0; bn < 4; ++bn)
      bfr[bn] = ld16(&Bt[bi][nsub + bn * 16 + nl][half * 16]);
#pragma unroll
    for (int am = 0; am < 2; ++am)
#pragma unroll
      for (int bn = 0; bn < 4; ++bn)
        acc[am][bn] = wmma_f16(afr[am], bfr[bn], acc[am][bn]);
  }

  // ---- epilogue: uniform branch hoisted out of the store loops ----
  if (!tmode) {
    // row-major f16 (q, k)
#pragma unroll
    for (int am = 0; am < 2; ++am)
#pragma unroll
      for (int bn = 0; bn < 4; ++bn)
#pragma unroll
        for (int j = 0; j < 8; ++j) {
          const int gm = m0 + msub + am * 16 + half * 8 + j;
          const int gn = n0 + nsub + bn * 16 + nl;
          dst[(size_t)gm * C_SZ + gn] = (_Float16)acc[am][bn][j];
        }
  } else {
    // transposed vT[b][d][t]: 8 consecutive t per fragment -> one v8h store
#pragma unroll
    for (int am = 0; am < 2; ++am)
#pragma unroll
      for (int bn = 0; bn < 4; ++bn) {
        v8h pack;
#pragma unroll
        for (int j = 0; j < 8; ++j) pack[j] = (_Float16)acc[am][bn][j];
        const int gm = m0 + msub + am * 16 + half * 8;   // row base (mult of 8)
        const int b  = gm >> 11, t = gm & (T_SZ - 1);
        const int gn = n0 + nsub + bn * 16 + nl;
        *(v8h*)(dst + (size_t)b * C_SZ * T_SZ + (size_t)gn * T_SZ + t) = pack;
      }
  }
}

// ---------- stage 2: flash attention ----------
// one workgroup (8 waves) per 16 query rows; wave w owns K-slice and V-column
// slice [w*128, w*128+128).
__global__ __launch_bounds__(256)
void attn_fwd(const _Float16* __restrict__ qh, const _Float16* __restrict__ kh,
              const _Float16* __restrict__ vT, float* __restrict__ out) {
  __shared__ float   Sp[8][16][32];   // per-wave partial scores
  __shared__ _Float16 P[16][32];      // softmax probs for current key block
  __shared__ float   alphaS[16];
  __shared__ float   linvS[16];

  const int tid  = threadIdx.x;
  const int wave = tid >> 5, lane = tid & 31;
  const int nl = lane & 15, half = lane >> 4;
  const int gm0 = blockIdx.x * 16;          // flattened query-row base
  const int b   = gm0 >> 11;
  const int t0  = gm0 & (T_SZ - 1);
  const _Float16* qb = qh + (size_t)b * T_SZ * C_SZ;
  const _Float16* kb = kh + (size_t)b * T_SZ * C_SZ;
  const _Float16* vb = vT + (size_t)b * C_SZ * T_SZ;

  // preload Q A-fragments for this wave's 128-wide K slice (4 k-steps)
  const int kbase = wave * 128;
  v16h qf[4];
#pragma unroll
  for (int s = 0; s < 4; ++s) {
    const _Float16* qp = qb + (size_t)(t0 + nl) * C_SZ + kbase + s * 32;
    qf[s] = mk16(*(const v8h*)(qp + half * 16),
                 *(const v8h*)(qp + half * 16 + 8));
  }

  v8f O[8] = {};                       // 16 rows x 128 V-cols (this wave)

  // softmax owner threads: 16 threads per row
  const int srow = tid >> 4, scol = tid & 15;
  float mRun = -INFINITY, lRun = 0.f;
  const float scale = 0.03125f;        // 1024^-0.5

  const int nblk = ((t0 + 15) >> 5) + 1;
  for (int blk = 0; blk < nblk; ++blk) {
    const int s0 = blk * 32;

    // prefetch next K/V block toward the WGP while we compute this one
    if (blk + 1 < nblk) {
      const int sn = s0 + 32;
      __builtin_prefetch(kb + (size_t)(sn + lane) * C_SZ + kbase, 0, 1);
      __builtin_prefetch(vb + (size_t)(kbase + lane * 4) * T_SZ + sn, 0, 1);
    }

    // ---- partial S = q_slice @ k_sliceᵀ (this wave's 128 K values) ----
    v8f sc[2] = {};
#pragma unroll
    for (int s = 0; s < 4; ++s)
#pragma unroll
      for (int nt = 0; nt < 2; ++nt) {
        const _Float16* kp = kb + (size_t)(s0 + nt * 16 + nl) * C_SZ +
                             kbase + s * 32 + half * 16;
        sc[nt] = wmma_f16(qf[s], ld16(kp), sc[nt]);
      }
#pragma unroll
    for (int nt = 0; nt < 2; ++nt)
#pragma unroll
      for (int j = 0; j < 8; ++j)
        Sp[wave][j + half * 8][nt * 16 + nl] = sc[nt][j];
    __syncthreads();

    // ---- cross-wave reduce + online softmax (16 threads per row) ----
    float sv0 = 0.f, sv1 = 0.f;
#pragma unroll
    for (int w = 0; w < 8; ++w) {
      sv0 += Sp[w][srow][scol];
      sv1 += Sp[w][srow][scol + 16];
    }
    const int tq = t0 + srow;
    float a0 = (s0 + scol      <= tq) ? sv0 * scale : -INFINITY;
    float a1 = (s0 + 16 + scol <= tq) ? sv1 * scale : -INFINITY;
    float bm = fmaxf(a0, a1);
#pragma unroll
    for (int off = 8; off; off >>= 1) bm = fmaxf(bm, __shfl_xor(bm, off, 16));
    const float mNew  = fmaxf(mRun, bm);
    const float p0    = __expf(a0 - mNew);   // exp(-inf - finite) == 0
    const float p1    = __expf(a1 - mNew);
    const float alpha = __expf(mRun - mNew); // first block: exp(-inf) == 0
    float ps = p0 + p1;
#pragma unroll
    for (int off = 8; off; off >>= 1) ps += __shfl_xor(ps, off, 16);
    lRun = alpha * lRun + ps;
    mRun = mNew;
    P[srow][scol]      = (_Float16)p0;
    P[srow][scol + 16] = (_Float16)p1;
    if (scol == 0) alphaS[srow] = alpha;
    __syncthreads();

    // ---- O = alpha*O + P @ V_block (this wave's 128 cols) ----
    float aj[8];
#pragma unroll
    for (int j = 0; j < 8; ++j) aj[j] = alphaS[j + half * 8];
    const _Float16* pp = &P[nl][0];
    v16h pf = mk16(*(const v8h*)(pp + half * 16),
                   *(const v8h*)(pp + half * 16 + 8));
#pragma unroll
    for (int nt = 0; nt < 8; ++nt) {
#pragma unroll
      for (int j = 0; j < 8; ++j) O[nt][j] *= aj[j];
      const _Float16* vp =
          vb + (size_t)(wave * 128 + nt * 16 + nl) * T_SZ + s0 + half * 16;
      O[nt] = wmma_f16(pf, ld16(vp), O[nt]);
    }
    __syncthreads();   // protect Sp / P before next block
  }

  // ---- epilogue: normalize by running sum and store fp32 ----
  if (scol == 0) linvS[srow] = 1.0f / lRun;
  __syncthreads();
  float lj[8];
#pragma unroll
  for (int j = 0; j < 8; ++j) lj[j] = linvS[j + half * 8];
  float* ob = out + (size_t)b * T_SZ * C_SZ;
#pragma unroll
  for (int nt = 0; nt < 8; ++nt)
#pragma unroll
    for (int j = 0; j < 8; ++j)
      ob[(size_t)(t0 + j + half * 8) * C_SZ + wave * 128 + nt * 16 + nl] =
          O[nt][j] * lj[j];
}

// ---------- launch ----------
extern "C" void kernel_launch(void* const* d_in, const int* in_sizes, int n_in,
                              void* d_out, int out_size, void* d_ws,
                              size_t ws_size, hipStream_t stream) {
  (void)in_sizes; (void)n_in; (void)out_size; (void)ws_size;
  const float* x  = (const float*)d_in[0];
  const float* Wq = (const float*)d_in[1];
  const float* Wk = (const float*)d_in[2];
  const float* Wv = (const float*)d_in[3];
  float* out = (float*)d_out;

  char* ws = (char*)d_ws;
  const size_t XH_BYTES = (size_t)M_TOTAL * C_SZ * 2;     // 16 MB
  const size_t W_BYTES  = (size_t)C_SZ * C_SZ * 2;        // 2 MB each
  _Float16* xh = (_Float16*)(ws);
  _Float16* Wh = (_Float16*)(ws + XH_BYTES);              // Wq|Wk|Wv
  _Float16* qh = (_Float16*)(ws + XH_BYTES + 3 * W_BYTES);
  _Float16* kh = (_Float16*)(ws + 2 * XH_BYTES + 3 * W_BYTES);
  _Float16* vT = (_Float16*)(ws + 3 * XH_BYTES + 3 * W_BYTES);

  const int nX = M_TOTAL * C_SZ;          // 8.39M
  const int nW = C_SZ * C_SZ;             // 1.05M
  cvt_f32_to_f16<<<4096, 256, 0, stream>>>(x, xh, nX);
  cvt_f32_to_f16<<<1024, 256, 0, stream>>>(Wq, Wh, nW);
  cvt_f32_to_f16<<<1024, 256, 0, stream>>>(Wk, Wh + (size_t)nW, nW);
  cvt_f32_to_f16<<<1024, 256, 0, stream>>>(Wv, Wh + (size_t)2 * nW, nW);

  dim3 g1(M_TOTAL / 128, C_SZ / 128, 3);  // (64, 8, 3)
  qkv_gemm<<<g1, 256, 0, stream>>>(xh, Wh, qh, kh, vT);

  attn_fwd<<<M_TOTAL / 16, 256, 0, stream>>>(qh, kh, vT, out);
}
